// Classifier_2284922601480
// MI455X (gfx1250) — matrix operand order repaired and checked by reading, not compiled, not run on hardware
//
#include <hip/hip_runtime.h>
#include <hip/hip_bf16.h>
#include <math.h>

// Problem constants (match reference)
#define B_    1024
#define C_    256
#define S_    64
#define D_    128
#define TOPK_ 8
#define EPS_  1e-8f

typedef __attribute__((ext_vector_type(16))) __bf16       v16bf;
typedef __attribute__((ext_vector_type(8)))  float        v8f;
typedef __attribute__((ext_vector_type(4)))  unsigned int u32x4;
typedef __attribute__((ext_vector_type(8)))  int          i32x8;
typedef __attribute__((ext_vector_type(4)))  int          i32x4;

static __device__ __forceinline__ unsigned short f2bf(float f) {
  unsigned int u = __float_as_uint(f);
  u += 0x7fffu + ((u >> 16) & 1u);          // round-to-nearest-even
  return (unsigned short)(u >> 16);
}
static __device__ __forceinline__ float bf2f(unsigned short h) {
  return __uint_as_float(((unsigned int)h) << 16);
}
// Fast hardware reciprocal (v_rcp_f32, 1 TRANS op instead of div expansion)
static __device__ __forceinline__ float frcp(float x) {
  return __builtin_amdgcn_rcpf(x);
}

// ---------------------------------------------------------------------------
// Prep: f32 -> bf16 weight tiles + L2 row norms for weight (C*S rows) and emb
// (B rows). One wave per 128-wide row: 4 floats/lane, shfl-xor reduction.
// ---------------------------------------------------------------------------
__global__ __launch_bounds__(256) void prep_kernel(
    const float* __restrict__ emb, const float* __restrict__ weight,
    unsigned short* __restrict__ wbf, float* __restrict__ n_w,
    float* __restrict__ n_e) {
  const int row  = blockIdx.x * 8 + (threadIdx.x >> 5);
  const int lane = threadIdx.x & 31;
  if (row < C_ * S_) {
    const float4 v = ((const float4*)(weight + (size_t)row * D_))[lane];
    float ssq = v.x * v.x + v.y * v.y + v.z * v.z + v.w * v.w;
#pragma unroll
    for (int m = 16; m >= 1; m >>= 1) ssq += __shfl_xor(ssq, m, 32);
    uint2 pk;
    pk.x = (unsigned int)f2bf(v.x) | ((unsigned int)f2bf(v.y) << 16);
    pk.y = (unsigned int)f2bf(v.z) | ((unsigned int)f2bf(v.w) << 16);
    ((uint2*)(wbf + (size_t)row * D_))[lane] = pk;
    if (lane == 0) n_w[row] = sqrtf(ssq);
  } else {
    const int r = row - C_ * S_;  // emb row, r < B_
    const float4 v = ((const float4*)(emb + (size_t)r * D_))[lane];
    float ssq = v.x * v.x + v.y * v.y + v.z * v.z + v.w * v.w;
#pragma unroll
    for (int m = 16; m >= 1; m >>= 1) ssq += __shfl_xor(ssq, m, 32);
    if (lane == 0) n_e[r] = sqrtf(ssq);
  }
}

// ---------------------------------------------------------------------------
// Main fused kernel: block = (16 b-rows) x (one cluster c), 128 threads.
//  1) TDM pulls the contiguous 16KB bf16 weight tile [64s x 128d] into LDS.
//  2) 4 waves each compute one 16(s) x 16(b) WMMA tile over K=128 (bf16).
//  3) scores -> LDS, top-8 + softmax per b, proto dot/norm, final score.
// ---------------------------------------------------------------------------
__global__ __launch_bounds__(128) void route_kernel(
    const float* __restrict__ emb, const unsigned short* __restrict__ wbf,
    const float* __restrict__ n_w, const float* __restrict__ n_e,
    float* __restrict__ out) {
  __shared__ __align__(16) unsigned short ldsW[S_ * D_];   // 16KB weight tile
  __shared__ __align__(16) float          ldsEf[16 * D_];  // 8KB f32 emb tile
  __shared__ __align__(16) unsigned short ldsEt[D_ * 16];  // 4KB bf16 emb [b][d]
  __shared__ __align__(16) float          sc[S_ * 16];     // scores [s][b]
  __shared__ float neS[16];
  __shared__ float nwS[S_];
  __shared__ float wkS[16][TOPK_];
  __shared__ short idxS[16][TOPK_];

  const int tid  = threadIdx.x;
  const int lane = tid & 31;
  const int wv   = tid >> 5;
  const int b0   = blockIdx.x * 16;
  const int c    = blockIdx.y;

  // ---- Tensor Data Mover: contiguous 16KB copy weight_bf16[c] -> ldsW ----
  if (tid == 0) {
    const unsigned long long ga =
        (unsigned long long)(const void*)(wbf + (size_t)c * S_ * D_);
    const unsigned int la = (unsigned int)(unsigned long long)(const void*)ldsW;
    u32x4 g0;
    g0[0] = 1u;                                   // count=1, user descriptor
    g0[1] = la;                                   // lds_addr
    g0[2] = (unsigned int)(ga & 0xffffffffull);   // global_addr[31:0]
    g0[3] = (unsigned int)((ga >> 32) & 0x01ffffffu) | (2u << 30);  // type=2
    i32x8 g1;
    g1[0] = (3 << 16);     // workgroup_mask=0, data_size=3 (8B elements)
    g1[1] = (2048 << 16);  // tensor_dim0 = 2048 (x8B = 16KB)
    g1[2] = (1 << 16);     // tensor_dim1 = 1
    g1[3] = (2048 << 16);  // tile_dim0 = 2048
    g1[4] = 0;             // tile_dim1 = tile_dim2 = 0 (unused)
    g1[5] = 2048;          // tensor_dim0_stride
    g1[6] = 0;
    g1[7] = 0;
    i32x4 z4 = {0, 0, 0, 0};
    i32x8 z8 = {0, 0, 0, 0, 0, 0, 0, 0};
    __builtin_amdgcn_tensor_load_to_lds(g0, g1, z4, z4, z8, 0);
  }

  // ---- overlap: stage emb tile (f32 + b-major bf16) and norms ----
  {
    const float4* src = (const float4*)(emb + (size_t)b0 * D_);
    float4*       dst = (float4*)ldsEf;
#pragma unroll
    for (int i = 0; i < 4; ++i) dst[tid + i * 128] = src[tid + i * 128];
    // b-major bf16 copy: thread tid == d, writes ldsEt[b][d]
#pragma unroll
    for (int b = 0; b < 16; ++b)
      ldsEt[b * D_ + tid] = f2bf(emb[(size_t)(b0 + b) * D_ + tid]);
    if (tid < 16) neS[tid] = n_e[b0 + tid];
    if (tid < S_) nwS[tid] = n_w[c * S_ + tid];
  }
  __builtin_amdgcn_s_wait_tensorcnt(0);
  __syncthreads();

  // ---- WMMA: M=s(16 per wave), N=b(16), K=d(128 = 4 steps of 32) ----
  const int half = lane >> 4;
  const int mrow = lane & 15;
  const int s0   = wv * 16;
  v8f acc = {};
#pragma unroll
  for (int ks = 0; ks < 4; ++ks) {
    union { v16bf v; unsigned int u[8]; } A, Bm;
    // A: weight rows, per-ISA lane layout (lanes 0-15: K 0-7,16-23; hi: +8)
    const unsigned int* wrow =
        (const unsigned int*)(ldsW + (size_t)(s0 + mrow) * D_);
#pragma unroll
    for (int p = 0; p < 8; ++p) {
      const int d = ks * 32 + 2 * p + 8 * half + ((p >= 4) ? 8 : 0);
      A.u[p] = wrow[d >> 1];
    }
    // B: emb rows (N=b=lane%16), same K layout as A
    const unsigned int* erow =
        (const unsigned int*)(ldsEt + (size_t)mrow * D_);
#pragma unroll
    for (int p = 0; p < 8; ++p) {
      const int d = ks * 32 + 2 * p + 8 * half + ((p >= 4) ? 8 : 0);
      Bm.u[p] = erow[d >> 1];
    }
    acc = __builtin_amdgcn_wmma_f32_16x16x32_bf16(
        false, A.v, false, Bm.v, (short)0, acc, false, false);
  }

  // ---- routing scores -> LDS (D layout: VGPR r -> M=r / r+8, N=lane%16) ----
  {
    const int   bcol = lane & 15;
    const float ne   = neS[bcol];
#pragma unroll
    for (int r = 0; r < 8; ++r) {
      const int   s    = s0 + r + 8 * half;
      const float cosv = acc[r] * frcp(fmaxf(ne * nwS[s], EPS_));
      sc[s * 16 + bcol] = fmaf(cosv, 0.5f, 0.5f) + EPS_;  // ROUTING_TEMP=1
    }
  }
  __syncthreads();

  // ---- top-8 of 64 + softmax per b (first-index tie-break like lax.top_k) --
  if (tid < 16) {
    const int b = tid;
    unsigned long long used = 0ull;
    float tk[TOPK_]; int ti[TOPK_];
#pragma unroll
    for (int k = 0; k < TOPK_; ++k) {
      float best = -1e30f; int bi = 0;
      for (int s = 0; s < S_; ++s) {
        if (!((used >> s) & 1ull)) {
          const float v = sc[s * 16 + b];
          if (v > best) { best = v; bi = s; }
        }
      }
      used |= (1ull << bi);
      tk[k] = best; ti[k] = bi;
    }
    const float mx = tk[0];
    float e[TOPK_], sum = 0.f;
#pragma unroll
    for (int k = 0; k < TOPK_; ++k) { e[k] = __expf(tk[k] - mx); sum += e[k]; }
    const float inv = frcp(sum);
#pragma unroll
    for (int k = 0; k < TOPK_; ++k) {
      wkS[b][k]  = e[k] * inv;
      idxS[b][k] = (short)ti[k];
    }
  }
  __syncthreads();

  // ---- proto = sum_k w_k * weight[c, idx_k, :]; second cosine; output ----
  {
    const int b = tid >> 3;       // 8 lanes per b, groups contiguous in-wave
    const int g = tid & 7;
    float w[TOPK_]; int ix[TOPK_];
#pragma unroll
    for (int k = 0; k < TOPK_; ++k) { w[k] = wkS[b][k]; ix[k] = idxS[b][k]; }
    float dot2 = 0.f, nrm = 0.f;
#pragma unroll
    for (int j = 0; j < 16; ++j) {
      const int d = g * 16 + j;
      float p = 0.f;
#pragma unroll
      for (int k = 0; k < TOPK_; ++k)
        p = fmaf(w[k], bf2f(ldsW[ix[k] * D_ + d]), p);
      dot2 = fmaf(p, ldsEf[b * D_ + d], dot2);
      nrm  = fmaf(p, p, nrm);
    }
#pragma unroll
    for (int m = 4; m >= 1; m >>= 1) {
      dot2 += __shfl_xor(dot2, m, 8);
      nrm  += __shfl_xor(nrm, m, 8);
    }
    if (g == 0) {
      const float cos2 = dot2 * frcp(fmaxf(sqrtf(nrm) * neS[b], EPS_));
      out[(size_t)(b0 + b) * C_ + c] = (fmaf(cos2, 0.5f, 0.5f) + EPS_) * 10.0f;
    }
  }
}

// ---------------------------------------------------------------------------
extern "C" void kernel_launch(void* const* d_in, const int* in_sizes, int n_in,
                              void* d_out, int out_size, void* d_ws,
                              size_t ws_size, hipStream_t stream) {
  (void)in_sizes; (void)n_in; (void)out_size; (void)ws_size;
  const float* emb    = (const float*)d_in[0];
  const float* weight = (const float*)d_in[1];
  // ws layout: bf16 weight (4MB) | n_w (64KB) | n_e (4KB)   (~4.26MB total)
  unsigned short* wbf = (unsigned short*)d_ws;
  float* n_w = (float*)((char*)d_ws + (size_t)C_ * S_ * D_ * 2);
  float* n_e = n_w + C_ * S_;
  float* out = (float*)d_out;

  // (C*S + B) rows = 17408 -> 8 waves/block -> 2176 blocks
  prep_kernel<<<dim3(2176), dim3(256), 0, stream>>>(emb, weight, wbf, n_w, n_e);
  route_kernel<<<dim3(B_ / 16, C_), dim3(128), 0, stream>>>(emb, wbf, n_w, n_e,
                                                            out);
}
